// SparseLinear_6365141533108
// MI455X (gfx1250) — compile-verified, matrix operand order checked by reading, not run on hardware
//
#include <hip/hip_runtime.h>

#define IN_CH  4096
#define OUT_CH 4096
#define BATCH  8192

#define BM 128
#define BN 256
#define BK 32
#define LDS_STRIDE 40   // 32 halfs + 8 pad -> 80B row stride, conflict-free b128 reads
#define NTILES (IN_CH / BK)

typedef __attribute__((ext_vector_type(16))) __bf16 v16bf;
typedef __attribute__((ext_vector_type(8)))  __bf16 v8bf;
typedef __attribute__((ext_vector_type(8)))  float  v8f;

// types matching the async-to-LDS builtin signature:
//   void __builtin_amdgcn_global_load_async_to_lds_b128(v4i AS1*, v4i AS3*, imm int, imm int)
typedef int v4i __attribute__((vector_size(4 * sizeof(int))));
typedef __attribute__((address_space(1))) v4i as1_v4i;
typedef __attribute__((address_space(3))) v4i as3_v4i;

#if __has_builtin(__builtin_amdgcn_global_load_async_to_lds_b128)
#define ASYNC_LOADS 1
#else
#define ASYNC_LOADS 0
#endif

__device__ __forceinline__ unsigned int f2bf(float f) {
    unsigned int u = __float_as_uint(f);
    u += 0x7FFFu + ((u >> 16) & 1u);   // round-to-nearest-even
    return u >> 16;
}

// one 16-byte global -> LDS transfer (async DMA when available)
__device__ __forceinline__ void ld_b128_to_lds(const unsigned short* g, unsigned short* l) {
#if ASYNC_LOADS
    __builtin_amdgcn_global_load_async_to_lds_b128(
        (as1_v4i*)g, (as3_v4i*)l, /*imm offset=*/0, /*cpol=*/0);
#else
    *(uint4*)l = *(const uint4*)g;
#endif
}

// wait for the CURRENT tile's async copies; the 6 async ops for the next tile
// may remain outstanding (async loads complete in order per wave)
__device__ __forceinline__ void wait_tile_async(bool next_in_flight) {
#if ASYNC_LOADS
#if __has_builtin(__builtin_amdgcn_s_wait_asynccnt)
    if (next_in_flight) __builtin_amdgcn_s_wait_asynccnt(6);
    else                __builtin_amdgcn_s_wait_asynccnt(0);
#else
    if (next_in_flight) asm volatile("s_wait_asynccnt 0x6" ::: "memory");
    else                asm volatile("s_wait_asynccnt 0x0" ::: "memory");
#endif
#endif
}

// cooperatively issue one K-tile: A = 128x32 halfs (512 chunks), B = 256x32 (1024 chunks)
// -> 2 + 4 = 6 b128 transfers per thread
__device__ __forceinline__ void issue_tile(const unsigned short* __restrict__ Xb,
                                           const unsigned short* __restrict__ Wd,
                                           unsigned short* __restrict__ As,
                                           unsigned short* __restrict__ Bs,
                                           int t, int m0, int n0, int k0) {
#pragma unroll
    for (int i = 0; i < 2; ++i) {
        int c   = t + (i << 8);        // 0..511
        int row = c >> 2;              // 0..127
        int cp8 = (c & 3) << 3;        // half offset of 16B chunk
        ld_b128_to_lds(Xb + (size_t)(m0 + row) * IN_CH + k0 + cp8,
                       As + row * LDS_STRIDE + cp8);
    }
#pragma unroll
    for (int i = 0; i < 4; ++i) {
        int c   = t + (i << 8);        // 0..1023
        int row = c >> 2;              // 0..255
        int cp8 = (c & 3) << 3;
        ld_b128_to_lds(Wd + (size_t)(n0 + row) * IN_CH + k0 + cp8,
                       Bs + row * LDS_STRIDE + cp8);
    }
}

// ---------------- kernel 1: zero the dense bf16 W ----------------
__global__ void zero16_kernel(uint4* __restrict__ p, int n16) {
    int i = blockIdx.x * blockDim.x + threadIdx.x;
    if (i < n16) p[i] = make_uint4(0u, 0u, 0u, 0u);
}

// ---------------- kernel 2: COO scatter -> dense bf16 W ----------------
__global__ void scatter_kernel(const float* __restrict__ w,
                               const int* __restrict__ rows,
                               const int* __restrict__ cols,
                               unsigned short* __restrict__ Wd, int nnz) {
    int i = blockIdx.x * blockDim.x + threadIdx.x;
    if (i < nnz) {
        Wd[(size_t)rows[i] * IN_CH + cols[i]] = (unsigned short)f2bf(w[i]);
    }
}

// ---------------- kernel 3: x f32 -> bf16 (8 elems / thread) ----------------
__global__ void cvt_x_kernel(const float4* __restrict__ x4,
                             uint4* __restrict__ xb4, long long n) {
    long long i = (long long)blockIdx.x * blockDim.x + threadIdx.x;
    if (i >= n) return;
    float4 a = x4[2 * i];
    float4 b = x4[2 * i + 1];
    uint4 o;
    o.x = f2bf(a.x) | (f2bf(a.y) << 16);
    o.y = f2bf(a.z) | (f2bf(a.w) << 16);
    o.z = f2bf(b.x) | (f2bf(b.y) << 16);
    o.w = f2bf(b.z) | (f2bf(b.w) << 16);
    xb4[i] = o;
}

// ---------------- kernel 4: bf16 WMMA GEMM: out = Xb @ Wd^T + bias ----------------
// Block: 256 threads = 8 waves. Block tile 128x256, wave tile 64x64 (4x4 of 16x16).
// Double-buffered LDS, async global->LDS copies.
__global__ __launch_bounds__(256)
void gemm_bf16_kernel(const unsigned short* __restrict__ Xb,
                      const unsigned short* __restrict__ Wd,
                      const float* __restrict__ bias,
                      float* __restrict__ out) {
    __shared__ __align__(16) unsigned short As[2][BM * LDS_STRIDE];  // 2 x 10240 B
    __shared__ __align__(16) unsigned short Bs[2][BN * LDS_STRIDE];  // 2 x 20480 B

    const int t    = threadIdx.x;
    const int lane = t & 31;
    const int wid  = t >> 5;
    const int wm   = wid & 1;      // wave row  (2 x 64 = 128)
    const int wn   = wid >> 1;     // wave col  (4 x 64 = 256)
    const int m0   = blockIdx.y * BM;
    const int n0   = blockIdx.x * BN;
    const int g    = lane >> 4;    // half-wave group (K-run select)
    const int rl   = lane & 15;    // row/col within 16

    v8f acc[4][4];
    const v8f vzero = {0.f, 0.f, 0.f, 0.f, 0.f, 0.f, 0.f, 0.f};
#pragma unroll
    for (int i = 0; i < 4; ++i)
#pragma unroll
        for (int j = 0; j < 4; ++j) acc[i][j] = vzero;

    // prologue: fill buffer 0
    issue_tile(Xb, Wd, As[0], Bs[0], t, m0, n0, 0);

    for (int kt = 0; kt < NTILES; ++kt) {
        const int cur = kt & 1;
        if (kt + 1 < NTILES) {
            issue_tile(Xb, Wd, As[cur ^ 1], Bs[cur ^ 1], t, m0, n0, (kt + 1) * BK);
            wait_tile_async(true);    // current tile resident; next may be in flight
        } else {
            wait_tile_async(false);
        }
        __syncthreads();

        const unsigned short* Ac = As[cur];
        const unsigned short* Bc = Bs[cur];
        const int koff = g << 3;      // lane's first K-run (halfs)

        v16bf af[4];
        v16bf bq[4];
#pragma unroll
        for (int i = 0; i < 4; ++i) {
            const unsigned short* p = &Ac[(wm * 64 + i * 16 + rl) * LDS_STRIDE + koff];
            v8bf lo = *(const v8bf*)p;          // K = 8g .. 8g+7
            v8bf hi = *(const v8bf*)(p + 16);   // K = 16+8g .. 16+8g+7
            af[i] = __builtin_shufflevector(lo, hi,
                    0,1,2,3,4,5,6,7,8,9,10,11,12,13,14,15);
        }
#pragma unroll
        for (int j = 0; j < 4; ++j) {
            const unsigned short* p = &Bc[(wn * 64 + j * 16 + rl) * LDS_STRIDE + koff];
            v8bf lo = *(const v8bf*)p;
            v8bf hi = *(const v8bf*)(p + 16);
            bq[j] = __builtin_shufflevector(lo, hi,
                    0,1,2,3,4,5,6,7,8,9,10,11,12,13,14,15);
        }
#pragma unroll
        for (int i = 0; i < 4; ++i)
#pragma unroll
            for (int j = 0; j < 4; ++j)
                acc[i][j] = __builtin_amdgcn_wmma_f32_16x16x32_bf16(
                    /*neg_a=*/false, af[i], /*neg_b=*/false, bq[j],
                    /*c_mod=*/(short)0, acc[i][j],
                    /*reuse_a=*/false, /*reuse_b=*/false);

        __syncthreads();   // everyone done reading 'cur' before it is refilled
    }

    // epilogue: C/D layout -> lane l holds (M = r + 8*(l>>4), N = l&15) in VGPR r
#pragma unroll
    for (int j = 0; j < 4; ++j) {
        const int col = n0 + wn * 64 + j * 16 + rl;
        const float bv = bias[col];
#pragma unroll
        for (int i = 0; i < 4; ++i) {
            const int mb = m0 + wm * 64 + i * 16 + g * 8;
#pragma unroll
            for (int r = 0; r < 8; ++r) {
                out[(size_t)(mb + r) * OUT_CH + col] = acc[i][j][r] + bv;
            }
        }
    }
}

extern "C" void kernel_launch(void* const* d_in, const int* in_sizes, int n_in,
                              void* d_out, int out_size, void* d_ws, size_t ws_size,
                              hipStream_t stream) {
    const float* x    = (const float*)d_in[0];
    const float* sw   = (const float*)d_in[1];
    const int*   rows = (const int*)d_in[2];
    const int*   cols = (const int*)d_in[3];
    const float* bias = (const float*)d_in[4];
    const int    nnz  = in_sizes[1];
    float* out = (float*)d_out;

    // workspace layout: [Xb: 8192*4096 bf16 = 64 MiB][Wd: 4096*4096 bf16 = 32 MiB]
    unsigned short* Xb = (unsigned short*)d_ws;
    unsigned short* Wd = (unsigned short*)((char*)d_ws + (size_t)BATCH * IN_CH * sizeof(unsigned short));

    // 1) zero dense W (re-done every call: scatter writes only nnz entries)
    const int n16 = (OUT_CH * IN_CH * 2) / 16;   // 2,097,152 uint4
    zero16_kernel<<<(n16 + 255) / 256, 256, 0, stream>>>((uint4*)Wd, n16);

    // 2) scatter COO weights into dense bf16 W
    scatter_kernel<<<(nnz + 255) / 256, 256, 0, stream>>>(sw, rows, cols, Wd, nnz);

    // 3) convert x f32 -> bf16
    const long long nx8 = (long long)BATCH * IN_CH / 8;   // 4,194,304
    cvt_x_kernel<<<(int)((nx8 + 255) / 256), 256, 0, stream>>>(
        (const float4*)x, (uint4*)Xb, nx8);

    // 4) dense bf16 WMMA GEMM with fused bias (double-buffered, async LDS fill)
    dim3 grid(OUT_CH / BN, BATCH / BM);   // (16, 64)
    gemm_bf16_kernel<<<grid, 256, 0, stream>>>(Xb, Wd, bias, out);
}
